// LSTMEDProtoModule_19464791785505
// MI455X (gfx1250) — compile-verified, matrix operand order checked by reading, not run on hardware
//
#include <hip/hip_runtime.h>

// ---------------------------------------------------------------------------
// LSTM encoder-decoder for MI455X (gfx1250).
//  - 32 workgroups (one per 16-row batch tile), 256 threads = 8 wave32 waves;
//    no cross-workgroup sync (batch rows are independent through the recurrence).
//  - Gate GEMMs: v_wmma_f32_16x16x32_f16, f32 accumulate.
//  - All weight B-fragments are LOOP-INVARIANT and held in VGPRs (160 regs/wave),
//    filled once per phase. Occupancy is 2 waves/SIMD regardless, so the large
//    register allocation is free on wave32 (up to 1024 VGPRs addressable).
//  - h tile double-buffered in LDS (f16, padded rows -> conflict-free);
//    c never leaves registers. One s_barrier per encoder timestep, two per decoder timestep.
// ---------------------------------------------------------------------------

typedef __attribute__((ext_vector_type(16))) _Float16 v16h;
typedef __attribute__((ext_vector_type(8)))  _Float16 v8h;
typedef __attribute__((ext_vector_type(8)))  float    v8f;

namespace {
constexpr int Bb = 512, Ss = 512, Ff = 32, Hh = 128;
constexpr int BT      = 16;          // batch rows per workgroup (WMMA M)
constexpr int NWAVE   = 8;           // wave w owns hidden cols [16w, 16w+16)
constexpr int THREADS = 32 * NWAVE;  // 256
constexpr int HPAD    = 136;         // h row pitch in halves (68 dwords -> no bank conflicts)
constexpr int XPAD    = 40;          // staged-x row pitch in halves
}  // namespace

__device__ __forceinline__ float fast_sigmoid(float x) {
    return 1.0f / (1.0f + __expf(-x));
}
__device__ __forceinline__ float fast_tanh(float x) {
    float e = __expf(2.0f * x);
    return (e - 1.0f) / (e + 1.0f);
}

// Gather one B fragment (W^T tile) straight from global f32 into registers.
// W row-major [Ntotal, Kdim]; B[k][n] = W[n][k].
// Assumed f16 B layout (dual of the documented 16-bit A layout):
//   lanes 0-15 : column n, K = kk*32 + {0..7, 16..23}
//   lanes 16-31: column n, K = kk*32 + {8..15, 24..31}
__device__ __forceinline__ v16h load_bfrag_global(const float* __restrict__ W, int Kdim,
                                                  int n, int kk, int hb) {
    const float* Wrow = W + (size_t)n * Kdim + kk * 32 + hb * 8;
    v16h b;
#pragma unroll
    for (int e = 0; e < 8; ++e) b[e] = (_Float16)Wrow[e];        // klocal hb*8+{0..7}
#pragma unroll
    for (int e = 0; e < 8; ++e) b[e + 8] = (_Float16)Wrow[e + 16];  // +{16..23}
    return b;
}

// Load a 16x32 f16 A fragment from an LDS row (rowk -> element [m][kk*32]).
__device__ __forceinline__ v16h load_afrag_lds(const _Float16* rowk, int hb) {
    v8h lo = *(const v8h*)(rowk + hb * 8);
    v8h up = *(const v8h*)(rowk + hb * 8 + 16);
    v16h a;
#pragma unroll
    for (int i = 0; i < 8; ++i) { a[i] = lo[i]; a[i + 8] = up[i]; }
    return a;
}

// Load a 16x32 A fragment from a global f32 row, converting to f16.
__device__ __forceinline__ v16h load_afrag_global_f32(const float* __restrict__ row, int hb) {
    const float* p = row + hb * 8;
    float4 f0 = *(const float4*)(p);
    float4 f1 = *(const float4*)(p + 4);
    float4 f2 = *(const float4*)(p + 16);
    float4 f3 = *(const float4*)(p + 20);
    v16h a;
    a[0] = (_Float16)f0.x;  a[1] = (_Float16)f0.y;  a[2]  = (_Float16)f0.z;  a[3]  = (_Float16)f0.w;
    a[4] = (_Float16)f1.x;  a[5] = (_Float16)f1.y;  a[6]  = (_Float16)f1.z;  a[7]  = (_Float16)f1.w;
    a[8] = (_Float16)f2.x;  a[9] = (_Float16)f2.y;  a[10] = (_Float16)f2.z;  a[11] = (_Float16)f2.w;
    a[12] = (_Float16)f3.x; a[13] = (_Float16)f3.y; a[14] = (_Float16)f3.z;  a[15] = (_Float16)f3.w;
    return a;
}

// h @ Whh^T contribution: 4 k-steps x 4 gate tiles = 16 WMMAs, B from registers.
__device__ __forceinline__ void gate_gemm_h(v8f acc[4], const v16h whhF[4][4],
                                            const _Float16* hrow, int hb) {
#pragma unroll
    for (int kk = 0; kk < 4; ++kk) {
        v16h a = load_afrag_lds(hrow + kk * 32, hb);
#pragma unroll
        for (int g = 0; g < 4; ++g) {
            acc[g] = __builtin_amdgcn_wmma_f32_16x16x32_f16(
                false, a, false, whhF[g][kk], (short)0, acc[g], false, false);
        }
    }
}

// Elementwise LSTM update; writes new h (f16) to hdst, keeps c in registers.
__device__ __forceinline__ void lstm_update(const v8f acc[4], const float bg[4],
                                            float cReg[8], _Float16* hdst,
                                            int wave, int l15, int hb) {
#pragma unroll
    for (int r = 0; r < 8; ++r) {
        const float iv = fast_sigmoid(acc[0][r] + bg[0]);
        const float fv = fast_sigmoid(acc[1][r] + bg[1]);
        const float gv = fast_tanh(acc[2][r] + bg[2]);
        const float ov = fast_sigmoid(acc[3][r] + bg[3]);
        const float cn = fv * cReg[r] + iv * gv;
        cReg[r] = cn;
        const float hn = ov * fast_tanh(cn);
        hdst[(size_t)(r + hb * 8) * HPAD + wave * 16 + l15] = (_Float16)hn;
    }
}

__global__ __launch_bounds__(THREADS)
void lstm_ed_kernel(const float* __restrict__ x,
                    const float* __restrict__ enc_Wih, const float* __restrict__ enc_Whh,
                    const float* __restrict__ enc_bih, const float* __restrict__ enc_bhh,
                    const float* __restrict__ dec_Wih, const float* __restrict__ dec_Whh,
                    const float* __restrict__ dec_bih, const float* __restrict__ dec_bhh,
                    const float* __restrict__ Wout,    const float* __restrict__ bout,
                    float* __restrict__ out /* [B,S,F] output, then [B,H] latent */) {
    __shared__ _Float16 HBUF[2 * BT * HPAD];   // double-buffered h tile (f16)
    __shared__ _Float16 XS[BT * XPAD];         // decoder cell-input stage (f16)

    const int tid  = threadIdx.x;
    const int lane = tid & 31;
    const int wave = tid >> 5;
    const int l15  = lane & 15;
    const int hb   = lane >> 4;
    const int b0   = blockIdx.x * BT;
    const int nCol = wave * 16 + l15;          // this lane's hidden/gate column

    // ---------------- encoder weights -> register fragments ----------------
    v16h whhF[4][4], wihF[4];
    float bg[4];
#pragma unroll
    for (int g = 0; g < 4; ++g) {
        const int n = g * Hh + nCol;
#pragma unroll
        for (int kk = 0; kk < 4; ++kk)
            whhF[g][kk] = load_bfrag_global(enc_Whh, Hh, n, kk, hb);
        wihF[g] = load_bfrag_global(enc_Wih, Ff, n, 0, hb);
        bg[g]   = enc_bih[n] + enc_bhh[n];
    }

    for (int i = tid; i < BT * HPAD; i += THREADS) HBUF[i] = (_Float16)0.0f;  // h0 = 0
    __syncthreads();

    float cReg[8];
#pragma unroll
    for (int r = 0; r < 8; ++r) cReg[r] = 0.0f;  // c0 = 0

    int cur = 0;

    // ------------------------------ encoder ------------------------------
    const float* xrow_base = x + ((size_t)(b0 + l15) * Ss) * Ff;
    for (int t = 0; t < Ss; ++t) {
        if (t + 1 < Ss)
            __builtin_prefetch(xrow_base + (size_t)(t + 1) * Ff, 0, 0);

        // Issue the global x load first ...
        v16h ax = load_afrag_global_f32(xrow_base + (size_t)t * Ff, hb);

        v8f acc[4] = {v8f{}, v8f{}, v8f{}, v8f{}};

        // ... h @ Whh^T runs from LDS/registers while x is in flight ...
        gate_gemm_h(acc, whhF, HBUF + (size_t)cur * BT * HPAD + (size_t)l15 * HPAD, hb);

        // ... then x @ Wih^T (K = F = 32, one k-step).
#pragma unroll
        for (int g = 0; g < 4; ++g) {
            acc[g] = __builtin_amdgcn_wmma_f32_16x16x32_f16(
                false, ax, false, wihF[g], (short)0, acc[g], false, false);
        }

        lstm_update(acc, bg, cReg, HBUF + (size_t)(cur ^ 1) * BT * HPAD, wave, l15, hb);
        __syncthreads();
        cur ^= 1;
    }

    // latent = encoder-final c : out[B*S*F + b*H + n]
    {
        float* latent = out + (size_t)Bb * Ss * Ff;
#pragma unroll
        for (int r = 0; r < 8; ++r)
            latent[(size_t)(b0 + r + hb * 8) * Hh + nCol] = cReg[r];
    }

    // -------------- swap register weights: encoder -> decoder --------------
#pragma unroll
    for (int g = 0; g < 4; ++g) {
        const int n = g * Hh + nCol;
#pragma unroll
        for (int kk = 0; kk < 4; ++kk)
            whhF[g][kk] = load_bfrag_global(dec_Whh, Hh, n, kk, hb);
        wihF[g] = load_bfrag_global(dec_Wih, Ff, n, 0, hb);
        bg[g]   = dec_bih[n] + dec_bhh[n];
    }
    v16h woutF[4] = {v16h{}, v16h{}, v16h{}, v16h{}};
    float boReg = 0.0f;
    if (wave < 2) {  // waves 0,1 own the two 16-wide output column tiles
#pragma unroll
        for (int kk = 0; kk < 4; ++kk)
            woutF[kk] = load_bfrag_global(Wout, Hh, nCol, kk, hb);
        boReg = bout[nCol];
    }

    // ------------------------------ decoder ------------------------------
    // Decoder starts from (h_enc, c_enc): h is in HBUF[cur] (f16), c in cReg.
    for (int st = 0; st < Ss; ++st) {
        // waves 0,1: out = h @ Wout^T + bout -> global output (time-flipped) + x-stage
        if (wave < 2) {
            v8f oacc = v8f{};
            const _Float16* hrow = HBUF + (size_t)cur * BT * HPAD + (size_t)l15 * HPAD;
#pragma unroll
            for (int kk = 0; kk < 4; ++kk) {
                v16h a = load_afrag_lds(hrow + kk * 32, hb);
                oacc = __builtin_amdgcn_wmma_f32_16x16x32_f16(
                    false, a, false, woutF[kk], (short)0, oacc, false, false);
            }
            const int tout = Ss - 1 - st;
#pragma unroll
            for (int r = 0; r < 8; ++r) {
                const int   m = r + hb * 8;
                const float v = oacc[r] + boReg;
                out[((size_t)(b0 + m) * Ss + tout) * Ff + nCol] = v;
                XS[(size_t)m * XPAD + nCol] = (_Float16)v;
            }
        }
        __syncthreads();  // out staged before it is consumed as the cell input

        v8f acc[4] = {v8f{}, v8f{}, v8f{}, v8f{}};

        // h @ Whh^T
        gate_gemm_h(acc, whhF, HBUF + (size_t)cur * BT * HPAD + (size_t)l15 * HPAD, hb);

        // out @ Wih^T (K = 32, from the LDS stage)
        {
            v16h a = load_afrag_lds(XS + (size_t)l15 * XPAD, hb);
#pragma unroll
            for (int g = 0; g < 4; ++g) {
                acc[g] = __builtin_amdgcn_wmma_f32_16x16x32_f16(
                    false, a, false, wihF[g], (short)0, acc[g], false, false);
            }
        }

        lstm_update(acc, bg, cReg, HBUF + (size_t)(cur ^ 1) * BT * HPAD, wave, l15, hb);
        __syncthreads();
        cur ^= 1;
    }
}

extern "C" void kernel_launch(void* const* d_in, const int* in_sizes, int n_in,
                              void* d_out, int out_size, void* d_ws, size_t ws_size,
                              hipStream_t stream) {
    (void)in_sizes; (void)n_in; (void)out_size; (void)d_ws; (void)ws_size;
    const float* x       = (const float*)d_in[0];
    const float* enc_Wih = (const float*)d_in[1];
    const float* enc_Whh = (const float*)d_in[2];
    const float* enc_bih = (const float*)d_in[3];
    const float* enc_bhh = (const float*)d_in[4];
    const float* dec_Wih = (const float*)d_in[5];
    const float* dec_Whh = (const float*)d_in[6];
    const float* dec_bih = (const float*)d_in[7];
    const float* dec_bhh = (const float*)d_in[8];
    const float* Wout    = (const float*)d_in[9];
    const float* boutp   = (const float*)d_in[10];
    float* out = (float*)d_out;

    lstm_ed_kernel<<<Bb / BT, THREADS, 0, stream>>>(
        x, enc_Wih, enc_Whh, enc_bih, enc_bhh,
        dec_Wih, dec_Whh, dec_bih, dec_bhh, Wout, boutp, out);
}
